// TransformerBlock_41704132444188
// MI455X (gfx1250) — matrix operand order repaired and checked by reading, not compiled, not run on hardware
//
#include <hip/hip_runtime.h>
#include <hip/hip_bf16.h>
#include <math.h>

#define BB 4
#define SL 2048
#define DIM 1024
#define NH 16
#define HDIM 64
#define NE 8
#define DFF 4096

typedef __attribute__((ext_vector_type(16))) __bf16 v16bf;
typedef __attribute__((ext_vector_type(8)))  float  v8f;
typedef __attribute__((ext_vector_type(2)))  __bf16 bf16x2;
typedef __attribute__((ext_vector_type(8)))  __bf16 bf16x8;

__device__ __forceinline__ __bf16 f2bf(float f) {
  unsigned u = __builtin_bit_cast(unsigned, f);
  unsigned r = u + 0x7FFFu + ((u >> 16) & 1u);
  unsigned short s = (unsigned short)(r >> 16);
  return __builtin_bit_cast(__bf16, s);
}

// ---- WMMA fragment helpers (CDNA5 16x16x32 bf16, wave32 layouts) ----
// A 16x32: lane m=lane&15, half h=lane>>4; VGPR j holds K = (j>=4?16:0)+8h+2(j&3)..+1
__device__ __forceinline__ v16bf load_frag_a(const __bf16* base, int ld) {
  const int lane = threadIdx.x & 31;
  const int m = lane & 15, hi = lane >> 4;
  v16bf a;
#pragma unroll
  for (int j = 0; j < 8; ++j) {
    const int k0 = ((j & 4) << 2) + hi * 8 + ((j & 3) << 1);
    bf16x2 p = *(const bf16x2*)(base + m * ld + k0);
    a[2 * j] = p.x; a[2 * j + 1] = p.y;
  }
  return a;
}
// B 32x16 from TRANSPOSED [N][K] storage: lane n=lane&15, half h; VGPR j holds K = (j>=4?16:0)+8h+2(j&3)..+1
__device__ __forceinline__ v16bf load_frag_bt(const __bf16* base, int ld) {
  const int lane = threadIdx.x & 31;
  const int n = lane & 15, hi = lane >> 4;
  v16bf b;
#pragma unroll
  for (int j = 0; j < 8; ++j) {
    const int k0 = ((j & 4) << 2) + hi * 8 + ((j & 3) << 1);
    bf16x2 p = *(const bf16x2*)(base + n * ld + k0);
    b[2 * j] = p.x; b[2 * j + 1] = p.y;
  }
  return b;
}
__device__ __forceinline__ v8f wmma_bf16(v16bf a, v16bf b, v8f c) {
  return __builtin_amdgcn_wmma_f32_16x16x32_bf16(false, a, false, b, (short)0, c, false, false);
}

// ---- LayerNorm (f32 in -> f32 + bf16 out) ----
__launch_bounds__(256)
__global__ void ln_kernel(const float* __restrict__ x, const float* __restrict__ gamma,
                          float* __restrict__ hf, __bf16* __restrict__ hb) {
  __shared__ float sred[16];
  __shared__ float smv[2];
  const int row = blockIdx.x;
  const int t = threadIdx.x;
  const float* xr = x + (size_t)row * DIM;
  float s = 0.f, s2 = 0.f;
  float v[4];
#pragma unroll
  for (int i = 0; i < 4; ++i) { v[i] = xr[t + i * 256]; s += v[i]; s2 += v[i] * v[i]; }
#pragma unroll
  for (int off = 16; off > 0; off >>= 1) { s += __shfl_xor(s, off, 32); s2 += __shfl_xor(s2, off, 32); }
  const int wid = t >> 5, lane = t & 31;
  if (lane == 0) { sred[wid] = s; sred[8 + wid] = s2; }
  __syncthreads();
  if (t == 0) {
    float ts = 0.f, ts2 = 0.f;
    for (int i = 0; i < 8; ++i) { ts += sred[i]; ts2 += sred[8 + i]; }
    float mu = ts / (float)DIM;
    float var = ts2 / (float)DIM - mu * mu;
    smv[0] = mu; smv[1] = rsqrtf(var + 1e-5f);
  }
  __syncthreads();
  const float mu = smv[0], rs = smv[1];
#pragma unroll
  for (int i = 0; i < 4; ++i) {
    const int d = t + i * 256;
    const float hv = gamma[d] * (v[i] - mu) * rs;
    hf[(size_t)row * DIM + d] = hv;
    hb[(size_t)row * DIM + d] = f2bf(hv);
  }
}

__global__ void cvt_kernel(const float* __restrict__ src, __bf16* __restrict__ dst, int n) {
  int i = blockIdx.x * blockDim.x + threadIdx.x;
  if (i < n) dst[i] = f2bf(src[i]);
}

// ---- Tiled bf16 WMMA GEMM: C[MxN] = A[MxK] * B[KxN] ----
#define BM 64
#define BN 128
#define BK 32
#define BKP 40  // padded K stride in LDS (80B rows: 16B aligned, conflict-spread)

// MODE 0: store f32   MODE 1: store bf16*scale   MODE 2: store f32 + residual
template<int MODE>
__launch_bounds__(256)
__global__ void gemm_kernel(const __bf16* __restrict__ A, const __bf16* __restrict__ Bm,
                            float* __restrict__ Cf, __bf16* __restrict__ Cb,
                            const float* __restrict__ res,
                            int M, int N, int K, float scale) {
  __shared__ __align__(16) __bf16 As[BM * BKP];
  __shared__ __align__(16) __bf16 Bs[BN * BKP];
  const int t = threadIdx.x;
  const int wid = t >> 5;
  const int bn0 = blockIdx.x * BN;
  const int bm0 = blockIdx.y * BM;
  const int wm = (wid >> 2) * 32;
  const int wn = (wid & 3) * 32;
  v8f acc[2][2] = {};
  const int arow = t >> 2, ac8 = (t & 3) * 8;
  const int bk = t >> 4, bc8 = (t & 15) * 8;
  for (int kt = 0; kt < K; kt += BK) {
    __syncthreads();
    bf16x8 av = *(const bf16x8*)(A + (size_t)(bm0 + arow) * K + kt + ac8);
    *(bf16x8*)(As + arow * BKP + ac8) = av;
#pragma unroll
    for (int hh = 0; hh < 2; ++hh) {
      const int kk = hh * 16 + bk;
      bf16x8 bv = *(const bf16x8*)(Bm + (size_t)(kt + kk) * N + bn0 + bc8);
#pragma unroll
      for (int i = 0; i < 8; ++i) Bs[(bc8 + i) * BKP + kk] = bv[i];  // transpose -> [N][K]
    }
    __syncthreads();
    if (kt + BK < K) {  // prefetch next tiles (global_prefetch_b8)
      __builtin_prefetch(A + (size_t)(bm0 + arow) * K + kt + BK + ac8, 0, 0);
      __builtin_prefetch(Bm + (size_t)(kt + BK + bk) * N + bn0 + bc8, 0, 0);
    }
    v16bf a0 = load_frag_a(As + (wm + 0) * BKP, BKP);
    v16bf a1 = load_frag_a(As + (wm + 16) * BKP, BKP);
    v16bf b0 = load_frag_bt(Bs + (wn + 0) * BKP, BKP);
    v16bf b1 = load_frag_bt(Bs + (wn + 16) * BKP, BKP);
    acc[0][0] = wmma_bf16(a0, b0, acc[0][0]);
    acc[0][1] = wmma_bf16(a0, b1, acc[0][1]);
    acc[1][0] = wmma_bf16(a1, b0, acc[1][0]);
    acc[1][1] = wmma_bf16(a1, b1, acc[1][1]);
  }
  const int lane = t & 31, n = lane & 15, hi = lane >> 4;
#pragma unroll
  for (int mt = 0; mt < 2; ++mt)
#pragma unroll
    for (int nt = 0; nt < 2; ++nt)
#pragma unroll
      for (int r = 0; r < 8; ++r) {
        const int row = bm0 + wm + mt * 16 + r + 8 * hi;
        const int col = bn0 + wn + nt * 16 + n;
        const size_t idx = (size_t)row * N + col;
        const float v = acc[mt][nt][r];
        if (MODE == 0)      Cf[idx] = v;
        else if (MODE == 1) Cb[idx] = f2bf(v * scale);
        else                Cf[idx] = v + res[idx];
      }
}

// ---- Flash attention: MQA, q pre-scaled, one 16-row q tile per wave ----
#define KSP 72
#define VSP 40
#define PSP 40

__launch_bounds__(256)
__global__ void attn_kernel(const __bf16* __restrict__ q, const __bf16* __restrict__ kv,
                            __bf16* __restrict__ concat) {
  __shared__ __align__(16) __bf16 Ks[32 * KSP];       // [seq][hd]
  __shared__ __align__(16) __bf16 Vs[HDIM * VSP];     // [hd][seq] (transposed)
  __shared__ __align__(16) __bf16 Ps[8][16 * PSP];    // per-wave prob tile
  const int t = threadIdx.x;
  const int wid = t >> 5;
  const int lane = t & 31, nn = lane & 15, hi = lane >> 4;
  const int b = blockIdx.x >> 8;
  const int tile = (blockIdx.x & 255) * 8 + wid;
  const int h = tile >> 7;
  const int q0 = (tile & 127) * 16;
  const __bf16* qbase = q + (size_t)(b * SL + q0) * DIM + h * HDIM;
  v16bf qa0 = load_frag_a(qbase, DIM);
  v16bf qa1 = load_frag_a(qbase + 32, DIM);
  v8f o[4] = {};
  float mrow[8], lrow[8];
#pragma unroll
  for (int r = 0; r < 8; ++r) { mrow[r] = -1e30f; lrow[r] = 0.f; }
  const int crow = t >> 3, cc8 = (t & 7) * 8;
  for (int kc = 0; kc < SL; kc += 32) {
    __syncthreads();
    const __bf16* src = kv + (size_t)(b * SL + kc + crow) * 128;
    bf16x8 kvv = *(const bf16x8*)(src + cc8);
    *(bf16x8*)(Ks + crow * KSP + cc8) = kvv;
    bf16x8 vvv = *(const bf16x8*)(src + 64 + cc8);
#pragma unroll
    for (int i = 0; i < 8; ++i) Vs[(cc8 + i) * VSP + crow] = vvv[i];
    __syncthreads();
    // scores: 16 rows x 32 cols, contraction over hd=64
    v8f sc[2] = {};
#pragma unroll
    for (int nt = 0; nt < 2; ++nt) {
      const __bf16* kb = Ks + (nt * 16) * KSP;
      v16bf kb0 = load_frag_bt(kb, KSP);
      v16bf kb1 = load_frag_bt(kb + 32, KSP);
      sc[nt] = wmma_bf16(qa0, kb0, sc[nt]);
      sc[nt] = wmma_bf16(qa1, kb1, sc[nt]);
    }
    // online softmax update per row (rows r+8*hi; reduce over 16 lanes of half)
    __bf16* pw = Ps[wid];
#pragma unroll
    for (int r = 0; r < 8; ++r) {
      float mx = fmaxf(sc[0][r], sc[1][r]);
#pragma unroll
      for (int off = 1; off < 16; off <<= 1) mx = fmaxf(mx, __shfl_xor(mx, off, 32));
      const float mnew = fmaxf(mrow[r], mx);
      const float alpha = __expf(mrow[r] - mnew);
      mrow[r] = mnew;
      const float p0 = __expf(sc[0][r] - mnew);
      const float p1 = __expf(sc[1][r] - mnew);
      float rs = p0 + p1;
#pragma unroll
      for (int off = 1; off < 16; off <<= 1) rs += __shfl_xor(rs, off, 32);
      lrow[r] = lrow[r] * alpha + rs;
#pragma unroll
      for (int nt = 0; nt < 4; ++nt) o[nt][r] *= alpha;
      const int prow = r + 8 * hi;
      pw[prow * PSP + nn] = f2bf(p0);
      pw[prow * PSP + 16 + nn] = f2bf(p1);
    }
    // P(16x32) @ V(32x64)
    v16bf pa = load_frag_a(pw, PSP);
#pragma unroll
    for (int nt = 0; nt < 4; ++nt) {
      v16bf vb = load_frag_bt(Vs + (nt * 16) * VSP, VSP);
      o[nt] = wmma_bf16(pa, vb, o[nt]);
    }
  }
#pragma unroll
  for (int nt = 0; nt < 4; ++nt)
#pragma unroll
    for (int r = 0; r < 8; ++r) {
      const int row = q0 + r + 8 * hi;
      const int col = h * HDIM + nt * 16 + nn;
      concat[(size_t)(b * SL + row) * DIM + col] = f2bf(o[nt][r] / lrow[r]);
    }
}

// ---- MoE router + experts (bandwidth-bound; VALU) ----
__launch_bounds__(256)
__global__ void logits_kernel(const float* __restrict__ m, const float* __restrict__ phi,
                              float* __restrict__ lg) {
  const int t = threadIdx.x;
  const int row = blockIdx.x * 16 + (t >> 4);
  const int e = t & 15;
  const float* mr = m + (size_t)row * DIM;
  float acc = 0.f;
  for (int d = 0; d < DIM; ++d) acc += mr[d] * phi[d * 16 + e];
  lg[(size_t)row * 16 + e] = acc;
}

__global__ void dispatch_kernel(const float* __restrict__ lg, float* __restrict__ dsp) {
  const int row = blockIdx.x * blockDim.x + threadIdx.x;
  if (row >= BB * SL) return;
  float v[16], mx = -1e30f;
#pragma unroll
  for (int i = 0; i < 16; ++i) { v[i] = lg[(size_t)row * 16 + i]; mx = fmaxf(mx, v[i]); }
  float s = 0.f;
#pragma unroll
  for (int i = 0; i < 16; ++i) { v[i] = __expf(v[i] - mx); s += v[i]; }
  const float inv = 1.f / s;
#pragma unroll
  for (int i = 0; i < 16; ++i) dsp[(size_t)row * 16 + i] = v[i] * inv;
}

__launch_bounds__(256)
__global__ void combine_kernel(const float* __restrict__ lg, float* __restrict__ cmb) {
  __shared__ float sred[8];
  __shared__ float sbc[2];
  const int b = blockIdx.x >> 4;
  const int e = blockIdx.x & 15;
  const int t = threadIdx.x;
  const float* base = lg + (size_t)b * SL * 16 + e;
  float mx = -1e30f;
  for (int s = t; s < SL; s += 256) mx = fmaxf(mx, base[(size_t)s * 16]);
#pragma unroll
  for (int off = 16; off > 0; off >>= 1) mx = fmaxf(mx, __shfl_xor(mx, off, 32));
  if ((t & 31) == 0) sred[t >> 5] = mx;
  __syncthreads();
  if (t == 0) { float m2 = -1e30f; for (int i = 0; i < 8; ++i) m2 = fmaxf(m2, sred[i]); sbc[0] = m2; }
  __syncthreads();
  mx = sbc[0];
  float sum = 0.f;
  for (int s = t; s < SL; s += 256) sum += __expf(base[(size_t)s * 16] - mx);
#pragma unroll
  for (int off = 16; off > 0; off >>= 1) sum += __shfl_xor(sum, off, 32);
  if ((t & 31) == 0) sred[t >> 5] = sum;
  __syncthreads();
  if (t == 0) { float s2 = 0.f; for (int i = 0; i < 8; ++i) s2 += sred[i]; sbc[1] = 1.f / s2; }
  __syncthreads();
  const float inv = sbc[1];
  float* out = cmb + (size_t)b * SL * 16 + e;
  for (int s = t; s < SL; s += 256) out[(size_t)s * 16] = __expf(base[(size_t)s * 16] - mx) * inv;
}

__launch_bounds__(256)
__global__ void xs_kernel(const float* __restrict__ dsp, const float* __restrict__ m,
                          float* __restrict__ xs) {
  const int dchunk = blockIdx.x & 3;
  const int slot = (blockIdx.x >> 2) & 15;
  const int b = blockIdx.x >> 6;
  const int d = dchunk * 256 + threadIdx.x;
  float acc = 0.f;
  const float* mb = m + (size_t)b * SL * DIM;
  const float* db = dsp + (size_t)b * SL * 16 + slot;
  for (int s = 0; s < SL; ++s) acc += db[(size_t)s * 16] * mb[(size_t)s * DIM + d];
  xs[((size_t)b * 16 + slot) * DIM + d] = acc;
}

__launch_bounds__(256)
__global__ void hid_kernel(const float* __restrict__ xs, const float* __restrict__ w1,
                           const float* __restrict__ b1, float* __restrict__ hid) {
  const int fc = blockIdx.x & 15;
  const int e = blockIdx.x >> 4;
  const int f = fc * 256 + threadIdx.x;
  float acc[8];
#pragma unroll
  for (int j = 0; j < 8; ++j) acc[j] = 0.f;
  const float* wb = w1 + (size_t)e * DIM * DFF + f;
  for (int d = 0; d < DIM; ++d) {
    const float w = wb[(size_t)d * DFF];
#pragma unroll
    for (int j = 0; j < 8; ++j)
      acc[j] += xs[((size_t)(j >> 1) * 16 + e * 2 + (j & 1)) * DIM + d] * w;
  }
  const float bias = b1[(size_t)e * DFF + f];
#pragma unroll
  for (int j = 0; j < 8; ++j) {
    const float xv = acc[j] + bias;
    const float g = 0.5f * xv * (1.f + erff(xv * 0.70710678118654752f));
    hid[(((size_t)(j >> 1) * NE + e) * 2 + (j & 1)) * DFF + f] = g;
  }
}

__launch_bounds__(256)
__global__ void ys_kernel(const float* __restrict__ hid, const float* __restrict__ w2,
                          const float* __restrict__ b2, float* __restrict__ ys) {
  const int dc = blockIdx.x & 3;
  const int e = blockIdx.x >> 2;
  const int d = dc * 256 + threadIdx.x;
  float acc[8];
#pragma unroll
  for (int j = 0; j < 8; ++j) acc[j] = 0.f;
  const float* wb = w2 + (size_t)e * DFF * DIM + d;
  for (int f = 0; f < DFF; ++f) {
    const float w = wb[(size_t)f * DIM];
#pragma unroll
    for (int j = 0; j < 8; ++j)
      acc[j] += hid[(((size_t)(j >> 1) * NE + e) * 2 + (j & 1)) * DFF + f] * w;
  }
  const float bias = b2[(size_t)e * DIM + d];
#pragma unroll
  for (int j = 0; j < 8; ++j)
    ys[((size_t)(j >> 1) * 16 + e * 2 + (j & 1)) * DIM + d] = acc[j] + bias;
}

__launch_bounds__(256)
__global__ void final_kernel(const float* __restrict__ cmb, const float* __restrict__ ys,
                             const float* __restrict__ m, float* __restrict__ out) {
  const int row = blockIdx.x;   // b*SL + s
  const int b = row >> 11;
  const int t = threadIdx.x;
  float c[16];
#pragma unroll
  for (int j = 0; j < 16; ++j) c[j] = cmb[(size_t)row * 16 + j];
#pragma unroll
  for (int i = 0; i < 4; ++i) {
    const int d = t + i * 256;
    float acc = m[(size_t)row * DIM + d];
#pragma unroll
    for (int j = 0; j < 16; ++j) acc += c[j] * ys[((size_t)b * 16 + j) * DIM + d];
    out[(size_t)row * DIM + d] = acc;
  }
}

extern "C" void kernel_launch(void* const* d_in, const int* in_sizes, int n_in,
                              void* d_out, int out_size, void* d_ws, size_t ws_size,
                              hipStream_t stream) {
  (void)in_sizes; (void)n_in; (void)out_size; (void)ws_size;
  const float* x   = (const float*)d_in[0];
  const float* wq  = (const float*)d_in[1];
  const float* wkv = (const float*)d_in[2];
  const float* wo  = (const float*)d_in[3];
  const float* g1  = (const float*)d_in[4];
  const float* g2  = (const float*)d_in[5];
  const float* phi = (const float*)d_in[6];
  const float* w1  = (const float*)d_in[7];
  const float* b1  = (const float*)d_in[8];
  const float* w2  = (const float*)d_in[9];
  const float* b2  = (const float*)d_in[10];
  float* out = (float*)d_out;

  char* ws = (char*)d_ws;
  size_t off = 0;
  auto alloc = [&](size_t bytes) -> void* {
    void* p = ws + off;
    off = (off + bytes + 255) & ~(size_t)255;
    return p;
  };
  const size_t ROWS = (size_t)BB * SL;          // 8192
  float*  h_f   = (float*) alloc(ROWS * DIM * 4);
  __bf16* h_b   = (__bf16*)alloc(ROWS * DIM * 2);
  __bf16* wq_b  = (__bf16*)alloc((size_t)DIM * DIM * 2);
  __bf16* wkv_b = (__bf16*)alloc((size_t)DIM * 128 * 2);
  __bf16* wo_b  = (__bf16*)alloc((size_t)DIM * DIM * 2);
  __bf16* q_b   = (__bf16*)alloc(ROWS * DIM * 2);
  __bf16* kv_b  = (__bf16*)alloc(ROWS * 128 * 2);
  __bf16* cat_b = (__bf16*)alloc(ROWS * DIM * 2);
  float*  x1_f  = (float*) alloc(ROWS * DIM * 4);
  float*  m_f   = (float*) alloc(ROWS * DIM * 4);
  __bf16* m_b   = (__bf16*)alloc(ROWS * DIM * 2);
  float*  lg    = (float*) alloc(ROWS * 16 * 4);
  float*  dsp   = (float*) alloc(ROWS * 16 * 4);
  float*  cmb   = (float*) alloc(ROWS * 16 * 4);
  float*  xs    = (float*) alloc((size_t)BB * 16 * DIM * 4);
  float*  hid   = (float*) alloc((size_t)BB * NE * 2 * DFF * 4);
  float*  ysb   = (float*) alloc((size_t)BB * 16 * DIM * 4);

  ln_kernel<<<(unsigned)ROWS, 256, 0, stream>>>(x, g1, h_f, h_b);
  cvt_kernel<<<(DIM * DIM + 255) / 256, 256, 0, stream>>>(wq, wq_b, DIM * DIM);
  cvt_kernel<<<(DIM * 128 + 255) / 256, 256, 0, stream>>>(wkv, wkv_b, DIM * 128);
  cvt_kernel<<<(DIM * DIM + 255) / 256, 256, 0, stream>>>(wo, wo_b, DIM * DIM);
  // q = h@wq, scaled by 1/sqrt(HD)=0.125, stored bf16
  gemm_kernel<1><<<dim3(DIM / BN, (unsigned)(ROWS / BM)), 256, 0, stream>>>(
      h_b, wq_b, nullptr, q_b, nullptr, (int)ROWS, DIM, DIM, 0.125f);
  // kv = h@wkv, stored bf16 (k | v)
  gemm_kernel<1><<<dim3(128 / BN, (unsigned)(ROWS / BM)), 256, 0, stream>>>(
      h_b, wkv_b, nullptr, kv_b, nullptr, (int)ROWS, 128, DIM, 1.0f);
  // flash attention -> concat (bf16)
  attn_kernel<<<BB * 256, 256, 0, stream>>>(q_b, kv_b, cat_b);
  // x1 = concat@wo + h
  gemm_kernel<2><<<dim3(DIM / BN, (unsigned)(ROWS / BM)), 256, 0, stream>>>(
      cat_b, wo_b, x1_f, nullptr, h_f, (int)ROWS, DIM, DIM, 1.0f);
  // m = layernorm(x1)
  ln_kernel<<<(unsigned)ROWS, 256, 0, stream>>>(x1_f, g2, m_f, m_b);
  // router
  logits_kernel<<<(unsigned)(ROWS / 16), 256, 0, stream>>>(m_f, phi, lg);
  dispatch_kernel<<<(unsigned)((ROWS + 255) / 256), 256, 0, stream>>>(lg, dsp);
  combine_kernel<<<BB * 16, 256, 0, stream>>>(lg, cmb);
  // xs[b,slot,:] = sum_s dispatch[b,s,slot] * m[b,s,:]
  xs_kernel<<<BB * 16 * (DIM / 256), 256, 0, stream>>>(dsp, m_f, xs);
  // experts (M=8 per expert -> weight-bandwidth bound, read w1/w2 exactly once)
  hid_kernel<<<NE * (DFF / 256), 256, 0, stream>>>(xs, w1, b1, hid);
  ys_kernel<<<NE * (DIM / 256), 256, 0, stream>>>(hid, w2, b2, ysb);
  // y = combine@ys + m
  final_kernel<<<(unsigned)ROWS, 256, 0, stream>>>(cmb, ysb, m_f, out);
}
